// HeteroSageGNN_38431367365388
// MI455X (gfx1250) — matrix operand order repaired and checked by reading, not compiled, not run on hardware
//
#include <hip/hip_runtime.h>

// ---------------------------------------------------------------------------
// HeteroSAGE GNN for MI455X (gfx1250, wave32, WMMA).
// Aggregations: fp32 global atomics (memory-bound, L2-resident feature tables).
// GEMMs: f16 inputs, f32 accumulate via v_wmma_f32_16x16x32_f16.
// ---------------------------------------------------------------------------

typedef __attribute__((ext_vector_type(16))) _Float16 v16h;
typedef __attribute__((ext_vector_type(8)))  _Float16 v8h;
typedef __attribute__((ext_vector_type(8)))  float    v8f;

#define NA 100000
#define NP 200000
#define NT 50000
#define DA 128
#define DP 128
#define DT 64
#define HCH 64
#define E_AP 400000
#define E_PA 400000
#define E_TP 300000
#define NEG_SLOPE 0.01f

static inline unsigned cdiv(long long a, long long b) { return (unsigned)((a + b - 1) / b); }

// ---- counts: cnt[dst[e]] += 1 ---------------------------------------------
__global__ void hsg_count_k(float* __restrict__ cnt, const int* __restrict__ dst, int E) {
  int e = blockIdx.x * blockDim.x + threadIdx.x;
  if (e < E) atomicAdd(&cnt[dst[e]], 1.0f);
}

// ---- scatter-add: agg[dst[e], :] += feat[src[e], :]  (4 floats / thread) ---
__global__ void hsg_scatter_k(float* __restrict__ agg, const float* __restrict__ feat,
                              const int* __restrict__ src, const int* __restrict__ dst,
                              int E, int chunkShift /* log2(D/4) */, int D) {
  long long t = (long long)blockIdx.x * blockDim.x + threadIdx.x;
  long long total = (long long)E << chunkShift;
  if (t >= total) return;
  int e = (int)(t >> chunkShift);
  int c = ((int)t & ((1 << chunkShift) - 1)) << 2;
  const float4 v = *(const float4*)(feat + (size_t)src[e] * D + c);
  float* p = agg + (size_t)dst[e] * D + c;
  atomicAdd(p + 0, v.x); atomicAdd(p + 1, v.y);
  atomicAdd(p + 2, v.z); atomicAdd(p + 3, v.w);
}

// ---- mean + convert to f16: out[i] = agg[i] / max(cnt[i>>dshift], 1) ------
__global__ void hsg_meancvt_k(_Float16* __restrict__ out, const float* __restrict__ agg,
                              const float* __restrict__ cnt, long long n, int dshift) {
  long long i = (long long)blockIdx.x * blockDim.x + threadIdx.x;
  if (i >= n) return;
  float c = fmaxf(cnt[i >> dshift], 1.0f);
  out[i] = (_Float16)(agg[i] / c);
}

// ---- plain f32 -> f16 convert ---------------------------------------------
__global__ void hsg_cvt_k(_Float16* __restrict__ out, const float* __restrict__ in, long long n) {
  long long i = (long long)blockIdx.x * blockDim.x + threadIdx.x;
  if (i < n) out[i] = (_Float16)in[i];
}

// ---- weight transpose+convert: WT[n*K+k] = (h)(W[k*64+n] (+W2[k*64+n])) ---
__global__ void hsg_wt_k(_Float16* __restrict__ WT, const float* __restrict__ W,
                         const float* __restrict__ W2, int K, int kshift) {
  int i = blockIdx.x * blockDim.x + threadIdx.x;
  if (i >= 64 * K) return;
  int n = i >> kshift, k = i & (K - 1);
  float v = W[k * 64 + n];
  if (W2) v += W2[k * 64 + n];
  WT[(size_t)n * K + k] = (_Float16)v;
}

// ---- bias broadcast init: out[i] = b1[i&63] (+ b2[i&63]) ------------------
__global__ void hsg_bias_k(float* __restrict__ out, const float* __restrict__ b1,
                           const float* __restrict__ b2, long long n) {
  long long i = (long long)blockIdx.x * blockDim.x + threadIdx.x;
  if (i >= n) return;
  float v = b1[i & 63];
  if (b2) v += b2[i & 63];
  out[i] = v;
}

// ---- LeakyReLU in place + f16 copy ----------------------------------------
__global__ void hsg_lrelu_k(float* __restrict__ buf, _Float16* __restrict__ outh, long long n) {
  long long i = (long long)blockIdx.x * blockDim.x + threadIdx.x;
  if (i >= n) return;
  float v = buf[i];
  v = v > 0.0f ? v : NEG_SLOPE * v;
  buf[i] = v;
  outh[i] = (_Float16)v;
}

// ---------------------------------------------------------------------------
// WMMA GEMM: C[M,64] += A[M,K](f16) * WT[64,K]^T(f16), f32 accumulate.
// 256 threads = 8 waves; wave w handles rows [base, base+16), all 64 cols.
// VGPR layouts per CDNA5 ISA 7.12.2 (16-bit A 16x32; B 32x16; f32 C 16x16).
// ---------------------------------------------------------------------------
__global__ void __launch_bounds__(256)
hsg_wmma_gemm_k(float* __restrict__ C, const _Float16* __restrict__ A,
                const _Float16* __restrict__ WT, int M, int K) {
  const int wave = threadIdx.x >> 5;               // 0..7 (wave-uniform)
  const int lane = threadIdx.x & 31;
  const int rowBase = (blockIdx.x * 8 + wave) * 16;
  if (rowBase >= M) return;                        // wave-uniform exit: EXEC stays all-1s

  const int l16  = lane & 15;
  const bool hi  = lane >= 16;
  const int rOff = hi ? 8 : 0;                     // C/D: lane>=16 holds rows M=8..15

  v8f acc[4];
#pragma unroll
  for (int nt = 0; nt < 4; ++nt)
#pragma unroll
    for (int r = 0; r < 8; ++r)
      acc[nt][r] = C[(size_t)(rowBase + rOff + r) * 64 + nt * 16 + l16];

  const _Float16* aRow = A + (size_t)(rowBase + l16) * K; // A: M = lane%16 for both halves
  const int aoff0 = hi ? 8 : 0;   // lane<16: K={0..7,16..23}; lane>=16: K={8..15,24..31}
  const int boff  = hi ? 8 : 0;   // B: lane<16 K={0..7,16..23}; lane>=16: K={8..15,24..31}

  for (int kc = 0; kc < K; kc += 32) {
    v8h alo = *(const v8h*)(aRow + kc + aoff0);
    v8h ahi = *(const v8h*)(aRow + kc + aoff0 + 16);
    v16h a;
#pragma unroll
    for (int i = 0; i < 8; ++i) { a[i] = alo[i]; a[i + 8] = ahi[i]; }

#pragma unroll
    for (int nt = 0; nt < 4; ++nt) {
      const _Float16* bRow = WT + (size_t)(nt * 16 + l16) * K + kc + boff;
      v8h b0 = *(const v8h*)(bRow);
      v8h b1 = *(const v8h*)(bRow + 16);
      v16h b;
#pragma unroll
      for (int i = 0; i < 8; ++i) { b[i] = b0[i]; b[i + 8] = b1[i]; }
      acc[nt] = __builtin_amdgcn_wmma_f32_16x16x32_f16(
          /*neg_a=*/false, a, /*neg_b=*/false, b,
          /*c_mod=*/(short)0, acc[nt], /*reuse_a=*/false, /*reuse_b=*/false);
    }
  }

#pragma unroll
  for (int nt = 0; nt < 4; ++nt)
#pragma unroll
    for (int r = 0; r < 8; ++r)
      C[(size_t)(rowBase + rOff + r) * 64 + nt * 16 + l16] = acc[nt][r];
}

// ---- final: out[r,0:4] = a2[r,:] @ lin_W[64,4] + lin_b --------------------
__global__ void hsg_final_k(float* __restrict__ out, const float* __restrict__ a2,
                            const float* __restrict__ W, const float* __restrict__ b, int M) {
  int r = blockIdx.x * blockDim.x + threadIdx.x;
  if (r >= M) return;
  float o0 = b[0], o1 = b[1], o2 = b[2], o3 = b[3];
  const float* row = a2 + (size_t)r * 64;
#pragma unroll 4
  for (int k = 0; k < 64; ++k) {
    float a = row[k];
    float4 w = *(const float4*)(W + 4 * k);
    o0 += a * w.x; o1 += a * w.y; o2 += a * w.z; o3 += a * w.w;
  }
  float4* po = (float4*)(out + (size_t)r * 4);
  *po = make_float4(o0, o1, o2, o3);
}

// ---------------------------------------------------------------------------
extern "C" void kernel_launch(void* const* d_in, const int* in_sizes, int n_in,
                              void* d_out, int out_size, void* d_ws, size_t ws_size,
                              hipStream_t stream) {
  // ---- inputs (setup_inputs order) ----
  const float* x_author = (const float*)d_in[0];
  const float* x_paper  = (const float*)d_in[1];
  const float* x_term   = (const float*)d_in[2];
  const int* src_ap = (const int*)d_in[3];
  const int* dst_ap = (const int*)d_in[4];
  const int* src_pa = (const int*)d_in[5];
  const int* dst_pa = (const int*)d_in[6];
  const int* src_tp = (const int*)d_in[7];
  const int* dst_tp = (const int*)d_in[8];
  const float* l1_ap_Wl = (const float*)d_in[11];
  const float* l1_ap_bl = (const float*)d_in[12];
  const float* l1_ap_Wr = (const float*)d_in[13];
  const float* l1_pa_Wl = (const float*)d_in[17];
  const float* l1_pa_bl = (const float*)d_in[18];
  const float* l1_pa_Wr = (const float*)d_in[19];
  const float* l2_pa_Wl = (const float*)d_in[20];
  const float* l2_pa_bl = (const float*)d_in[21];
  const float* l2_pa_Wr = (const float*)d_in[22];
  const float* l1_tp_Wl = (const float*)d_in[23];
  const float* l1_tp_Wr = (const float*)d_in[25];
  const float* lin_W = (const float*)d_in[35];
  const float* lin_b = (const float*)d_in[36];
  float* out = (float*)d_out;

  // ---- workspace carving (256B aligned) ----
  char* w = (char*)d_ws;
  auto alloc = [&](size_t bytes) -> void* {
    void* p = (void*)w;
    w += (bytes + 255) & ~(size_t)255;
    return p;
  };
  // zeroed region (contiguous): aggregates + counts
  char* zbeg = w;
  float* agg_ap = (float*)alloc((size_t)NP * DA * 4);
  float* cnt_ap = (float*)alloc((size_t)NP * 4);
  float* agg_tp = (float*)alloc((size_t)NP * DT * 4);
  float* cnt_tp = (float*)alloc((size_t)NP * 4);
  float* agg_pa = (float*)alloc((size_t)NA * DP * 4);
  float* cnt_pa = (float*)alloc((size_t)NA * 4);   // reused for layer-2 (same pa edges)
  float* agg2   = (float*)alloc((size_t)NA * HCH * 4);
  char* zend = w;
  // non-zeroed f32
  float* p1 = (float*)alloc((size_t)NP * HCH * 4);
  float* a1 = (float*)alloc((size_t)NA * HCH * 4);
  float* a2 = (float*)alloc((size_t)NA * HCH * 4);
  // f16 staging
  _Float16* mean_ap_h = (_Float16*)alloc((size_t)NP * DA * 2);
  _Float16* mean_tp_h = (_Float16*)alloc((size_t)NP * DT * 2);
  _Float16* mean_pa_h = (_Float16*)alloc((size_t)NA * DP * 2);
  _Float16* mean2_h   = (_Float16*)alloc((size_t)NA * HCH * 2);
  _Float16* xpaper_h  = (_Float16*)alloc((size_t)NP * DP * 2);
  _Float16* xauthor_h = (_Float16*)alloc((size_t)NA * DA * 2);
  _Float16* p1_h      = (_Float16*)alloc((size_t)NP * HCH * 2);
  _Float16* a1_h      = (_Float16*)alloc((size_t)NA * HCH * 2);
  // transposed f16 weights WT[64,K]
  _Float16* WT_ap_l1Wl = (_Float16*)alloc((size_t)64 * 128 * 2);
  _Float16* WT_Wr_p    = (_Float16*)alloc((size_t)64 * 128 * 2); // ap_Wr + tp_Wr (both act on x_paper)
  _Float16* WT_tp_l1Wl = (_Float16*)alloc((size_t)64 * 64 * 2);
  _Float16* WT_pa_l1Wl = (_Float16*)alloc((size_t)64 * 128 * 2);
  _Float16* WT_pa_l1Wr = (_Float16*)alloc((size_t)64 * 128 * 2);
  _Float16* WT_l2Wl    = (_Float16*)alloc((size_t)64 * 64 * 2);
  _Float16* WT_l2Wr    = (_Float16*)alloc((size_t)64 * 64 * 2);

  const int T = 256;

  // 0) zero aggregates/counts
  hipMemsetAsync(zbeg, 0, (size_t)(zend - zbeg), stream);

  // 1) weight prep
  hsg_wt_k<<<cdiv(64 * 128, T), T, 0, stream>>>(WT_ap_l1Wl, l1_ap_Wl, nullptr, 128, 7);
  hsg_wt_k<<<cdiv(64 * 128, T), T, 0, stream>>>(WT_Wr_p,    l1_ap_Wr, l1_tp_Wr, 128, 7);
  hsg_wt_k<<<cdiv(64 * 64,  T), T, 0, stream>>>(WT_tp_l1Wl, l1_tp_Wl, nullptr, 64, 6);
  hsg_wt_k<<<cdiv(64 * 128, T), T, 0, stream>>>(WT_pa_l1Wl, l1_pa_Wl, nullptr, 128, 7);
  hsg_wt_k<<<cdiv(64 * 128, T), T, 0, stream>>>(WT_pa_l1Wr, l1_pa_Wr, nullptr, 128, 7);
  hsg_wt_k<<<cdiv(64 * 64,  T), T, 0, stream>>>(WT_l2Wl,    l2_pa_Wl, nullptr, 64, 6);
  hsg_wt_k<<<cdiv(64 * 64,  T), T, 0, stream>>>(WT_l2Wr,    l2_pa_Wr, nullptr, 64, 6);

  // 2) f16 copies of root features
  hsg_cvt_k<<<cdiv((long long)NP * DP, T), T, 0, stream>>>(xpaper_h, x_paper, (long long)NP * DP);
  hsg_cvt_k<<<cdiv((long long)NA * DA, T), T, 0, stream>>>(xauthor_h, x_author, (long long)NA * DA);

  // 3) counts
  hsg_count_k<<<cdiv(E_AP, T), T, 0, stream>>>(cnt_ap, dst_ap, E_AP);
  hsg_count_k<<<cdiv(E_TP, T), T, 0, stream>>>(cnt_tp, dst_tp, E_TP);
  hsg_count_k<<<cdiv(E_PA, T), T, 0, stream>>>(cnt_pa, dst_pa, E_PA);

  // 4) layer-1 scatter-adds
  hsg_scatter_k<<<cdiv((long long)E_AP * 32, T), T, 0, stream>>>(agg_ap, x_author, src_ap, dst_ap, E_AP, 5, DA);
  hsg_scatter_k<<<cdiv((long long)E_TP * 16, T), T, 0, stream>>>(agg_tp, x_term,   src_tp, dst_tp, E_TP, 4, DT);
  hsg_scatter_k<<<cdiv((long long)E_PA * 32, T), T, 0, stream>>>(agg_pa, x_paper,  src_pa, dst_pa, E_PA, 5, DP);

  // 5) means -> f16
  hsg_meancvt_k<<<cdiv((long long)NP * DA, T), T, 0, stream>>>(mean_ap_h, agg_ap, cnt_ap, (long long)NP * DA, 7);
  hsg_meancvt_k<<<cdiv((long long)NP * DT, T), T, 0, stream>>>(mean_tp_h, agg_tp, cnt_tp, (long long)NP * DT, 6);
  hsg_meancvt_k<<<cdiv((long long)NA * DP, T), T, 0, stream>>>(mean_pa_h, agg_pa, cnt_pa, (long long)NA * DP, 7);

  // 6) bias init
  hsg_bias_k<<<cdiv((long long)NP * HCH, T), T, 0, stream>>>(p1, l1_ap_bl, d_in[24] ? (const float*)d_in[24] : nullptr, (long long)NP * HCH); // bl_ap + bl_tp
  hsg_bias_k<<<cdiv((long long)NA * HCH, T), T, 0, stream>>>(a1, l1_pa_bl, nullptr, (long long)NA * HCH);

  // 7) layer-1 WMMA GEMMs (accumulate through C across sequential launches)
  hsg_wmma_gemm_k<<<cdiv(NP, 128), T, 0, stream>>>(p1, mean_ap_h, WT_ap_l1Wl, NP, 128);
  hsg_wmma_gemm_k<<<cdiv(NP, 128), T, 0, stream>>>(p1, xpaper_h,  WT_Wr_p,    NP, 128);
  hsg_wmma_gemm_k<<<cdiv(NP, 128), T, 0, stream>>>(p1, mean_tp_h, WT_tp_l1Wl, NP, 64);
  hsg_wmma_gemm_k<<<cdiv(NA, 128), T, 0, stream>>>(a1, mean_pa_h, WT_pa_l1Wl, NA, 128);
  hsg_wmma_gemm_k<<<cdiv(NA, 128), T, 0, stream>>>(a1, xauthor_h, WT_pa_l1Wr, NA, 128);

  // 8) LeakyReLU (+ f16 copies)
  hsg_lrelu_k<<<cdiv((long long)NP * HCH, T), T, 0, stream>>>(p1, p1_h, (long long)NP * HCH);
  hsg_lrelu_k<<<cdiv((long long)NA * HCH, T), T, 0, stream>>>(a1, a1_h, (long long)NA * HCH);

  // 9) layer-2 aggregation over the same pa edges (reuse cnt_pa)
  hsg_scatter_k<<<cdiv((long long)E_PA * 16, T), T, 0, stream>>>(agg2, p1, src_pa, dst_pa, E_PA, 4, HCH);
  hsg_meancvt_k<<<cdiv((long long)NA * HCH, T), T, 0, stream>>>(mean2_h, agg2, cnt_pa, (long long)NA * HCH, 6);

  // 10) layer-2 GEMMs
  hsg_bias_k<<<cdiv((long long)NA * HCH, T), T, 0, stream>>>(a2, l2_pa_bl, nullptr, (long long)NA * HCH);
  hsg_wmma_gemm_k<<<cdiv(NA, 128), T, 0, stream>>>(a2, mean2_h, WT_l2Wl, NA, 64);
  hsg_wmma_gemm_k<<<cdiv(NA, 128), T, 0, stream>>>(a2, a1_h,    WT_l2Wr, NA, 64);

  // 11) final linear [NA,64] @ [64,4] + b
  hsg_final_k<<<cdiv(NA, T), T, 0, stream>>>(out, a2, lin_W, lin_b, NA);
}
